// MultiHeadAttention_29008209118034
// MI455X (gfx1250) — matrix operand order repaired and checked by reading, not compiled, not run on hardware
//
#include <hip/hip_runtime.h>
#include <hip/hip_bf16.h>

typedef _Float16 v16h __attribute__((ext_vector_type(16)));
typedef _Float16 h8   __attribute__((ext_vector_type(8)));
typedef float    v8f  __attribute__((ext_vector_type(8)));
typedef unsigned int u32x4 __attribute__((ext_vector_type(4)));
typedef int          i32x4 __attribute__((ext_vector_type(4)));
typedef int          i32x8 __attribute__((ext_vector_type(8)));

#define B_  4
#define S_  2048
#define D_  1024
#define H_  16
#define DK_ 64

// ---- WMMA operand builders -------------------------------------------------
// A operand (16x32 f16): lane holds row M=lane%16; elements 0-7 = K 8*half+[0..7],
// elements 8-15 = K 16+8*half+[0..7]  -> two contiguous 16B chunks.
static __device__ __forceinline__ v16h make_op16(const _Float16* c0,
                                                 const _Float16* c1) {
  h8 lo = *(const h8*)c0;
  h8 hi = *(const h8*)c1;
  v16h a;
#pragma unroll
  for (int i = 0; i < 8; ++i) { a[i] = lo[i]; a[i + 8] = hi[i]; }
  return a;
}
// B operand (32x16 f16): lane holds col N=lane%16; same K order as A:
// p = base + 8*half -> two 16B chunks at p, p+16.
static __device__ __forceinline__ v16h make_b16(const _Float16* p) {
  return make_op16(p, p + 16);
}

// ---- DPP16 butterfly reductions over the 16 lanes of each half -------------
template <int CTRL>
static __device__ __forceinline__ float dpp_xfer(float x) {
  return __int_as_float(__builtin_amdgcn_update_dpp(
      0, __float_as_int(x), CTRL, 0xF, 0xF, true));
}
static __device__ __forceinline__ float rowmax16(float x) {
  x = fmaxf(x, dpp_xfer<0xB1>(x));   // quad_perm [1,0,3,2]  (xor 1)
  x = fmaxf(x, dpp_xfer<0x4E>(x));   // quad_perm [2,3,0,1]  (xor 2)
  x = fmaxf(x, dpp_xfer<0x141>(x));  // row_half_mirror      (xor 7)
  x = fmaxf(x, dpp_xfer<0x140>(x));  // row_mirror           (xor 15)
  return x;
}
static __device__ __forceinline__ float rowsum16(float x) {
  x += dpp_xfer<0xB1>(x);
  x += dpp_xfer<0x4E>(x);
  x += dpp_xfer<0x141>(x);
  x += dpp_xfer<0x140>(x);
  return x;
}

// ---- TDM: DMA a 32x64 f16 tile into LDS with 72-half padded rows -----------
static __device__ __forceinline__ void tdm_load_k32x64(const _Float16* gsrc,
                                                       _Float16* ldst) {
  const unsigned lds_addr = (unsigned)(uintptr_t)ldst;
  const unsigned long long ga = (unsigned long long)(uintptr_t)gsrc;
  u32x4 g0 = {1u,                                       // count=1, user D#
              lds_addr,
              (unsigned)ga,                              // global_addr[31:0]
              (unsigned)((ga >> 32) & 0x1FFFFFFu) | (2u << 30)};  // ga[56:32], type=2
  // data_size=2B, pad_enable, pad_interval=32 dwords, pad_amount=4 dwords
  i32x8 g1 = {0x07110000,
              (int)(64u << 16),    // tensor_dim0 = 64
              (int)(2048u << 16),  // tensor_dim1 = S
              (int)(64u << 16),    // tile_dim0 = 64
              32,                  // tile_dim1 = 32
              64,                  // tensor_dim0_stride = 64
              0, 0};
  i32x4 gz = {0, 0, 0, 0};
#if __has_include(<hip/amd_detail/amd_gfx1250_TDM.h>)
  i32x8 gz8 = {0, 0, 0, 0, 0, 0, 0, 0};
  __builtin_amdgcn_tensor_load_to_lds(g0, g1, gz, gz, gz8, 0);
#else
  __builtin_amdgcn_tensor_load_to_lds(g0, g1, gz, gz, 0);
#endif
}

// ---- f32 -> f16 pre-conversion (8 elements/thread) -------------------------
__global__ __launch_bounds__(256) void cvt_f32_to_f16(
    const float* __restrict__ in, _Float16* __restrict__ out, int n8) {
  const int i = blockIdx.x * 256 + threadIdx.x;
  if (i < n8) {
    const float4* p = (const float4*)in + (size_t)i * 2;
    const float4 x = p[0], y = p[1];
    h8 o;
    o[0] = (_Float16)x.x; o[1] = (_Float16)x.y;
    o[2] = (_Float16)x.z; o[3] = (_Float16)x.w;
    o[4] = (_Float16)y.x; o[5] = (_Float16)y.y;
    o[6] = (_Float16)y.z; o[7] = (_Float16)y.w;
    ((h8*)out)[i] = o;
  }
}

// ---- GEMM: Y = X @ W^T, f16 inputs, LDS-staged 64x64 weight tile -----------
// Wave tile 32x64 (8 WMMAs per 32-k step), block tile 128x64.
// MODE 0: Y f16 scattered to [B,H,S,DK];  MODE 1: Y f32 [M,D] + bias
#define WLDS 72  // padded row stride (halfs) for the 64x64 weight tile
template <int MODE>
__global__ __launch_bounds__(128) void gemm_xwT(
    const _Float16* __restrict__ X, const _Float16* __restrict__ W16,
    const float* __restrict__ bias, void* __restrict__ Yv) {
  __shared__ _Float16 ldsW[64 * WLDS];

  const int lane  = threadIdx.x & 31;
  const int wv    = threadIdx.x >> 5;
  const int l16   = lane & 15;
  const int half  = lane >> 4;
  const int mbase = blockIdx.x * 128 + wv * 32;
  const int nbase = blockIdx.y * 64;
  const _Float16* Xrow0 = X + (size_t)(mbase + l16) * D_;
  const _Float16* Xrow1 = X + (size_t)(mbase + 16 + l16) * D_;

  // staging role: thread c covers weight row n=c>>1, k-chunk (c&1)*32
  const int sn = threadIdx.x >> 1;
  const int sk = (threadIdx.x & 1) * 32;
  const _Float16* Wsrc = W16 + (size_t)(nbase + sn) * D_ + sk;

  v8f acc[2][4];
#pragma unroll
  for (int mt = 0; mt < 2; ++mt)
#pragma unroll
    for (int t = 0; t < 4; ++t)
      acc[mt][t] = (v8f){0.f, 0.f, 0.f, 0.f, 0.f, 0.f, 0.f, 0.f};

  for (int kb = 0; kb < D_; kb += 64) {
#pragma unroll
    for (int cc = 0; cc < 4; ++cc)
      *(h8*)&ldsW[sn * WLDS + sk + cc * 8] = *(const h8*)(Wsrc + kb + cc * 8);
    __syncthreads();
#pragma unroll
    for (int ks = 0; ks < 2; ++ks) {
      const int ko = kb + ks * 32;
      v16h a0 = make_op16(Xrow0 + ko + half * 8, Xrow0 + ko + 16 + half * 8);
      v16h a1 = make_op16(Xrow1 + ko + half * 8, Xrow1 + ko + 16 + half * 8);
#pragma unroll
      for (int t = 0; t < 4; ++t) {
        v16h b = make_b16(&ldsW[(t * 16 + l16) * WLDS + ks * 32 + half * 8]);
        acc[0][t] = __builtin_amdgcn_wmma_f32_16x16x32_f16(
            false, a0, false, b, (short)0, acc[0][t], false, false);
        acc[1][t] = __builtin_amdgcn_wmma_f32_16x16x32_f16(
            false, a1, false, b, (short)0, acc[1][t], false, false);
      }
    }
    __syncthreads();
  }

#pragma unroll
  for (int mt = 0; mt < 2; ++mt) {
    const int mb = mbase + mt * 16;
    if (MODE == 0) {
      _Float16* Y = (_Float16*)Yv;
#pragma unroll
      for (int t = 0; t < 4; ++t) {
        const int o = nbase + t * 16 + l16;
        const int h = o >> 6, dk = o & 63;
#pragma unroll
        for (int r = 0; r < 8; ++r) {
          const int m = mb + r + 8 * half;
          const int b = m >> 11, s = m & (S_ - 1);
          Y[((size_t)(b * H_ + h) * S_ + s) * DK_ + dk] = (_Float16)acc[mt][t][r];
        }
      }
    } else {
      float* Y = (float*)Yv;
#pragma unroll
      for (int t = 0; t < 4; ++t) {
        const int o    = nbase + t * 16 + l16;
        const float bo = bias[o];
#pragma unroll
        for (int r = 0; r < 8; ++r) {
          const int m = mb + r + 8 * half;
          Y[(size_t)m * D_ + o] = acc[mt][t][r] + bo;
        }
      }
    }
  }
}

// ---- Flash attention with double-buffered TDM K tiles ----------------------
#define KLDS 72  // K tile row stride in halfs (TDM pad: 32 dwords + 4 dwords)
#define VLDS 40  // V^T tile row stride in halfs
#define PLDS 40  // P tile row stride in halfs
__global__ __launch_bounds__(128) void attn_kernel(
    const _Float16* __restrict__ Qp, const _Float16* __restrict__ Kp,
    const _Float16* __restrict__ Vp, _Float16* __restrict__ Ctx) {
  __shared__ _Float16 ldsK[2][32 * KLDS];
  __shared__ _Float16 ldsVt[64 * VLDS];
  __shared__ _Float16 ldsP[4][16 * PLDS];

  const int lane  = threadIdx.x & 31;
  const int wv    = threadIdx.x >> 5;
  const int l16   = lane & 15;
  const int half  = lane >> 4;
  const int qblk  = blockIdx.x;
  const int bh    = blockIdx.y;
  const int b     = bh >> 4, h = bh & 15;
  const int qbase = qblk * 64 + wv * 16;
  const int qmaxw = qbase + 15;
  const size_t base = (size_t)bh * S_ * DK_;

  // Q A-operands, pre-scaled by 1/sqrt(DK) = 0.125
  const _Float16* Qrow = Qp + base + (size_t)(qbase + l16) * DK_;
  v16h Qa[2];
#pragma unroll
  for (int st = 0; st < 2; ++st) {
    Qa[st] = make_op16(Qrow + st * 32 + half * 8, Qrow + st * 32 + 16 + half * 8);
#pragma unroll
    for (int i = 0; i < 16; ++i) Qa[st][i] = Qa[st][i] * (_Float16)0.125f;
  }

  v8f O[4];
#pragma unroll
  for (int t = 0; t < 4; ++t) O[t] = (v8f){0.f, 0.f, 0.f, 0.f, 0.f, 0.f, 0.f, 0.f};
  float mrun[8], lrun[8];
#pragma unroll
  for (int r = 0; r < 8; ++r) { mrun[r] = -1e30f; lrun[r] = 0.f; }

  const int nkt = qblk * 2 + 2;
  if (wv == 0) tdm_load_k32x64(Kp + base, &ldsK[0][0]);  // prefetch tile 0

  for (int kt = 0; kt < nkt; ++kt) {
    const int kk = kt * 32;
    __syncthreads();  // previous tile fully consumed

    if (wv == 0 && kt + 1 < nkt)  // prefetch next K tile while we stage/compute
      tdm_load_k32x64(Kp + base + (size_t)(kk + 32) * DK_, &ldsK[(kt + 1) & 1][0]);

    {  // cooperative V^T staging (overlaps the in-flight TDM transfers)
      const int tid = threadIdx.x;
#pragma unroll
      for (int j = 0; j < 2; ++j) {
        const int c    = tid + 128 * j;
        const int krow = c >> 3;
        const int doff = (c & 7) * 8;
        h8 vvec = *(const h8*)(Vp + base + (size_t)(kk + krow) * DK_ + doff);
#pragma unroll
        for (int e = 0; e < 8; ++e) ldsVt[(doff + e) * VLDS + krow] = vvec[e];
      }
    }
    if (wv == 0) {  // oldest TDM (tile kt) must be complete; newer may fly
      if (kt + 1 < nkt) __builtin_amdgcn_s_wait_tensorcnt((short)1);
      else              __builtin_amdgcn_s_wait_tensorcnt((short)0);
    }
    __syncthreads();

    if (kk <= qmaxw) {
      const _Float16* Kt = &ldsK[kt & 1][0];
      v8f s[2];
#pragma unroll
      for (int nt = 0; nt < 2; ++nt) {
        const int keyrow = nt * 16 + l16;
        v8f sc = (v8f){0.f, 0.f, 0.f, 0.f, 0.f, 0.f, 0.f, 0.f};
#pragma unroll
        for (int st = 0; st < 2; ++st) {
          v16h kb = make_b16(&Kt[keyrow * KLDS + st * 32 + half * 8]);
          sc = __builtin_amdgcn_wmma_f32_16x16x32_f16(false, Qa[st], false, kb,
                                                      (short)0, sc, false, false);
        }
        s[nt] = sc;
      }
      if (kk + 31 > qbase) {  // diagonal tile (wave-uniform): apply causal mask
#pragma unroll
        for (int nt = 0; nt < 2; ++nt) {
          const int col = kk + nt * 16 + l16;
#pragma unroll
          for (int r = 0; r < 8; ++r) {
            const int qrow = qbase + r + 8 * half;
            s[nt][r] = (col <= qrow) ? s[nt][r] : -1e30f;
          }
        }
      }
      float pr0[8], pr1[8];
#pragma unroll
      for (int r = 0; r < 8; ++r) {
        const float tmax  = rowmax16(fmaxf(s[0][r], s[1][r]));
        const float mnew  = fmaxf(mrun[r], tmax);
        const float alpha = __expf(mrun[r] - mnew);
        const float p0 = __expf(s[0][r] - mnew);
        const float p1 = __expf(s[1][r] - mnew);
        lrun[r] = lrun[r] * alpha + rowsum16(p0 + p1);
        mrun[r] = mnew;
        pr0[r] = p0; pr1[r] = p1;
#pragma unroll
        for (int t = 0; t < 4; ++t) O[t][r] *= alpha;
      }
      _Float16* P = &ldsP[wv][0];
#pragma unroll
      for (int r = 0; r < 8; ++r) {
        const int prow = r + 8 * half;
        P[prow * PLDS + l16]      = (_Float16)pr0[r];
        P[prow * PLDS + 16 + l16] = (_Float16)pr1[r];
      }
      asm volatile("s_wait_dscnt 0" ::: "memory");
      v16h Pa = make_op16(&P[l16 * PLDS + half * 8],
                          &P[l16 * PLDS + 16 + half * 8]);
#pragma unroll
      for (int t = 0; t < 4; ++t) {
        const int dv = t * 16 + l16;
        v16h vb = make_b16(&ldsVt[dv * VLDS + half * 8]);
        O[t] = __builtin_amdgcn_wmma_f32_16x16x32_f16(false, Pa, false, vb,
                                                      (short)0, O[t], false, false);
      }
    }
  }

#pragma unroll
  for (int r = 0; r < 8; ++r) {
    const float inv = 1.f / lrun[r];
    const int srow = qbase + r + 8 * half;
    _Float16* Crow = Ctx + ((size_t)b * S_ + srow) * D_ + h * DK_;
#pragma unroll
    for (int t = 0; t < 4; ++t)
      Crow[t * 16 + l16] = (_Float16)(O[t][r] * inv);
  }
}

extern "C" void kernel_launch(void* const* d_in, const int* in_sizes, int n_in,
                              void* d_out, int out_size, void* d_ws, size_t ws_size,
                              hipStream_t stream) {
  (void)in_sizes; (void)n_in; (void)out_size; (void)ws_size;
  const float* q  = (const float*)d_in[0];
  const float* k  = (const float*)d_in[1];
  const float* v  = (const float*)d_in[2];
  // d_in[3] = causal mask (structure known, unused)
  const float* Wq = (const float*)d_in[4];
  const float* Wk = (const float*)d_in[5];
  const float* Wv = (const float*)d_in[6];
  const float* Wo = (const float*)d_in[7];
  const float* bo = (const float*)d_in[8];
  float* out = (float*)d_out;

  const size_t NE = (size_t)B_ * S_ * D_;  // 8M elements
  const size_t MM = (size_t)D_ * D_;       // 1M elements
  _Float16* qh   = (_Float16*)d_ws;
  _Float16* kh   = qh + NE;
  _Float16* vh   = kh + NE;
  _Float16* Wq16 = vh + NE;
  _Float16* Wk16 = Wq16 + MM;
  _Float16* Wv16 = Wk16 + MM;
  _Float16* Wo16 = Wv16 + MM;
  _Float16* Qp   = Wo16 + MM;
  _Float16* Kp   = Qp + NE;
  _Float16* Vp   = Kp + NE;
  _Float16* Ctx  = Vp + NE;

  // pre-convert activations + weights to f16
  cvt_f32_to_f16<<<(int)(NE / 8 / 256), 256, 0, stream>>>(q, qh, (int)(NE / 8));
  cvt_f32_to_f16<<<(int)(NE / 8 / 256), 256, 0, stream>>>(k, kh, (int)(NE / 8));
  cvt_f32_to_f16<<<(int)(NE / 8 / 256), 256, 0, stream>>>(v, vh, (int)(NE / 8));
  cvt_f32_to_f16<<<(int)(MM / 8 / 256), 256, 0, stream>>>(Wq, Wq16, (int)(MM / 8));
  cvt_f32_to_f16<<<(int)(MM / 8 / 256), 256, 0, stream>>>(Wk, Wk16, (int)(MM / 8));
  cvt_f32_to_f16<<<(int)(MM / 8 / 256), 256, 0, stream>>>(Wv, Wv16, (int)(MM / 8));
  cvt_f32_to_f16<<<(int)(MM / 8 / 256), 256, 0, stream>>>(Wo, Wo16, (int)(MM / 8));

  const dim3 gg((B_ * S_) / 128, D_ / 64), bb(128);
  gemm_xwT<0><<<gg, bb, 0, stream>>>(qh, Wq16, nullptr, Qp);
  gemm_xwT<0><<<gg, bb, 0, stream>>>(kh, Wk16, nullptr, Kp);
  gemm_xwT<0><<<gg, bb, 0, stream>>>(vh, Wv16, nullptr, Vp);
  attn_kernel<<<dim3(S_ / 64, B_ * H_), bb, 0, stream>>>(Qp, Kp, Vp, Ctx);
  gemm_xwT<1><<<gg, bb, 0, stream>>>(Ctx, Wo16, bo, out);
}
